// UMT5Attention_29764123361587
// MI455X (gfx1250) — matrix operand order repaired and checked by reading, not compiled, not run on hardware
//
#include <hip/hip_runtime.h>
#include <hip/hip_bf16.h>
#include <math.h>

#define SEQ 2048
#define DM  2048
#define NH  32
#define DK  64

typedef unsigned short u16;
typedef __attribute__((ext_vector_type(16))) __bf16 v16bf;
typedef __attribute__((ext_vector_type(8)))  float  v8f;
typedef __attribute__((ext_vector_type(8)))  unsigned short u16x8;
typedef int v4i_g __attribute__((vector_size(16)));   // matches builtin's <4 x i32> pointee

// ---- gfx1250 async global->LDS path (probe via __has_builtin; fallback = sync) ----
#if defined(__has_builtin)
#  if __has_builtin(__builtin_amdgcn_global_load_async_to_lds_b128)
#    define HAVE_ASYNC_LDS 1
#  else
#    define HAVE_ASYNC_LDS 0
#  endif
#  if __has_builtin(__builtin_amdgcn_s_wait_asynccnt)
#    define ASYNC_WAIT0() __builtin_amdgcn_s_wait_asynccnt(0)
#  else
#    define ASYNC_WAIT0() asm volatile("s_wait_asynccnt 0" ::: "memory")
#  endif
#else
#  define HAVE_ASYNC_LDS 0
#  define ASYNC_WAIT0()
#endif

#define GAS(p) ((__attribute__((address_space(1))) v4i_g*)(p))
#define LAS(p) ((__attribute__((address_space(3))) v4i_g*)(p))

struct u16x8x2 { u16x8 lo; u16x8 hi; };

static __device__ __forceinline__ v16bf make_frag(u16x8 lo, u16x8 hi) {
    u16x8x2 t{lo, hi};
    return __builtin_bit_cast(v16bf, t);
}

static __device__ __forceinline__ u16 f2bf(float f) {
    unsigned u = __float_as_uint(f);
    u += 0x7FFFu + ((u >> 16) & 1u);   // round-to-nearest-even
    return (u16)(u >> 16);
}

static __device__ __forceinline__ float rmax16(float x) {
    x = fmaxf(x, __shfl_xor(x, 1));
    x = fmaxf(x, __shfl_xor(x, 2));
    x = fmaxf(x, __shfl_xor(x, 4));
    x = fmaxf(x, __shfl_xor(x, 8));
    return x;
}
static __device__ __forceinline__ float rsum16(float x) {
    x += __shfl_xor(x, 1);
    x += __shfl_xor(x, 2);
    x += __shfl_xor(x, 4);
    x += __shfl_xor(x, 8);
    return x;
}

// ---------------------------------------------------------------- cast fp32 -> bf16
__global__ __launch_bounds__(256) void cast_bf16_kernel(const float* __restrict__ in,
                                                        u16* __restrict__ out, int n) {
    for (int i = blockIdx.x * blockDim.x + threadIdx.x; i < n; i += gridDim.x * blockDim.x)
        out[i] = f2bf(in[i]);
}

// ------------------------------------------- transpose-cast: Wt[n][k] = bf16(W[k][n])
__global__ __launch_bounds__(256) void tcast_kernel(const float* __restrict__ W,
                                                    u16* __restrict__ Wt) {
    __shared__ float tile[32][33];
    int bx = blockIdx.x * 32, by = blockIdx.y * 32;
    int tx = threadIdx.x, ty = threadIdx.y;
#pragma unroll
    for (int i = 0; i < 32; i += 8)
        tile[ty + i][tx] = W[(by + ty + i) * DM + bx + tx];
    __syncthreads();
#pragma unroll
    for (int i = 0; i < 32; i += 8)
        Wt[(bx + ty + i) * DM + by + tx] = f2bf(tile[tx][ty + i]);
}

// ---------------------------------------------------- T5 relative position bias table
// biasTab[h][(k - q) + 2048] = rel_emb[bucket(k - q)][h]
__global__ __launch_bounds__(256) void bias_table_kernel(const float* __restrict__ rel_emb,
                                                         float* __restrict__ biasTab) {
    int idx = blockIdx.x * blockDim.x + threadIdx.x;   // 0..4095
    int h   = blockIdx.y;
    int rel = idx - 2048;                               // rel = k - q
    int n   = -rel;
    int ret = (n < 0) ? 16 : 0;
    n = (n < 0) ? -n : n;
    int b;
    if (n < 8) {
        b = n;
    } else {
        float t = __logf((float)n * 0.125f) * (8.0f / __logf(16.0f));
        b = 8 + (int)t;
        if (b > 15) b = 15;
    }
    biasTab[h * 4096 + idx] = rel_emb[(b + ret) * NH + h];
}

// -------------------------------------------------------------- WMMA bf16 GEMM
// C[2048x2048] = A(row-major bf16) x B (given transposed: Bt[n][k] bf16)
// Double-buffered LDS tiles filled with GLOBAL_LOAD_ASYNC_TO_LDS_B128 when available.
// MODE 0: store bf16 head-major  out[(col>>6)][row][col&63]   (Q, K)
// MODE 1: store fp32 row-major   out[row][col]                (final projection)
// MODE 2: store bf16 transposed  out[col][row]                (V -> Vt)
template <int MODE>
__global__ __launch_bounds__(256) void wmma_gemm_kernel(const u16* __restrict__ A,
                                                        const u16* __restrict__ Bt,
                                                        void* __restrict__ Out) {
    __shared__ u16 As[2][128][40];
    __shared__ u16 Bs[2][128][40];

    const int tid  = threadIdx.x;
    const int lane = tid & 31, wid = tid >> 5;
    const int wm = wid & 3, wn = wid >> 2;         // 4x2 wave grid -> 32M x 64N per wave
    const int ml = lane & 15, hf = lane >> 4;
    const int bM = blockIdx.y * 128, bN = blockIdx.x * 128;
    const int lr = tid >> 1, seg = tid & 1;        // tile-fill: row, 16-element segment

    const u16* gA = A  + (size_t)(bM + lr) * DM + seg * 16;
    const u16* gB = Bt + (size_t)(bN + lr) * DM + seg * 16;

    v8f acc[2][4];
#pragma unroll
    for (int mt = 0; mt < 2; ++mt)
#pragma unroll
        for (int nt = 0; nt < 4; ++nt)
            acc[mt][nt] = v8f{0.f, 0.f, 0.f, 0.f, 0.f, 0.f, 0.f, 0.f};

    constexpr int NT = DM / 32;

#if HAVE_ASYNC_LDS
    auto issue_tile = [&](int buf, int k0) {
        __builtin_amdgcn_global_load_async_to_lds_b128(
            GAS(gA + k0),     LAS(&As[buf][lr][seg * 16]),     0, 0);
        __builtin_amdgcn_global_load_async_to_lds_b128(
            GAS(gA + k0 + 8), LAS(&As[buf][lr][seg * 16 + 8]), 0, 0);
        __builtin_amdgcn_global_load_async_to_lds_b128(
            GAS(gB + k0),     LAS(&Bs[buf][lr][seg * 16]),     0, 0);
        __builtin_amdgcn_global_load_async_to_lds_b128(
            GAS(gB + k0 + 8), LAS(&Bs[buf][lr][seg * 16 + 8]), 0, 0);
    };
    issue_tile(0, 0);
#endif

    for (int kt = 0; kt < NT; ++kt) {
        const int k0  = kt * 32;
        const int buf = kt & 1;

#if HAVE_ASYNC_LDS
        ASYNC_WAIT0();            // my async copies for buf have landed in LDS
        __syncthreads();          // everyone's copies have landed; buf^1 consumers done
        if (kt + 1 < NT) issue_tile(buf ^ 1, k0 + 32);   // prefetch next tile async
#else
        __syncthreads();
        *(u16x8*)&As[buf][lr][seg * 16]     = *(const u16x8*)(gA + k0);
        *(u16x8*)&As[buf][lr][seg * 16 + 8] = *(const u16x8*)(gA + k0 + 8);
        *(u16x8*)&Bs[buf][lr][seg * 16]     = *(const u16x8*)(gB + k0);
        *(u16x8*)&Bs[buf][lr][seg * 16 + 8] = *(const u16x8*)(gB + k0 + 8);
        __syncthreads();
#endif

        v16bf af[2], bfr[4];
#pragma unroll
        for (int mt = 0; mt < 2; ++mt) {
            int row = wm * 32 + mt * 16 + ml;
            af[mt] = make_frag(*(const u16x8*)&As[buf][row][hf * 8],
                               *(const u16x8*)&As[buf][row][16 + hf * 8]);
        }
#pragma unroll
        for (int nt = 0; nt < 4; ++nt) {
            int row = wn * 64 + nt * 16 + ml;
            bfr[nt] = make_frag(*(const u16x8*)&Bs[buf][row][hf * 8],
                                *(const u16x8*)&Bs[buf][row][16 + hf * 8]);
        }
#pragma unroll
        for (int mt = 0; mt < 2; ++mt)
#pragma unroll
            for (int nt = 0; nt < 4; ++nt)
                acc[mt][nt] = __builtin_amdgcn_wmma_f32_16x16x32_bf16(
                    false, af[mt], false, bfr[nt], (short)0, acc[mt][nt], false, false);
    }

#pragma unroll
    for (int mt = 0; mt < 2; ++mt)
#pragma unroll
        for (int nt = 0; nt < 4; ++nt) {
            int col = bN + wn * 64 + nt * 16 + ml;
#pragma unroll
            for (int v = 0; v < 8; ++v) {
                int   row = bM + wm * 32 + mt * 16 + v + 8 * hf;
                float x   = acc[mt][nt][v];
                if (MODE == 0) {
                    ((u16*)Out)[(size_t)(col >> 6) * (SEQ * DK) + (size_t)row * DK + (col & 63)] = f2bf(x);
                } else if (MODE == 1) {
                    ((float*)Out)[(size_t)row * DM + col] = x;
                } else {
                    ((u16*)Out)[(size_t)col * SEQ + row] = f2bf(x);
                }
            }
        }
}

// -------------------------------------------------------------- flash attention
// Qb,Kb: bf16 [h][s][d] ; Vt: bf16 [h*64+d][s] ; biasTab: fp32 [h][4096]
// Ob: bf16 [s][h*64+d]
__global__ __launch_bounds__(128) void flash_attn_kernel(const u16* __restrict__ Qb,
                                                         const u16* __restrict__ Kb,
                                                         const u16* __restrict__ Vt,
                                                         const float* __restrict__ biasTab,
                                                         u16* __restrict__ Ob) {
    __shared__ u16 Pb[4][16][72];

    const int tid  = threadIdx.x;
    const int lane = tid & 31, wid = tid >> 5;
    const int ml = lane & 15, hf = lane >> 4;
    const int h     = blockIdx.y;
    const int qbase = blockIdx.x * 64 + wid * 16;

    // Q fragments (A layout), d split in two 32-chunks
    v16bf qf[2];
    {
        const u16* qp = Qb + ((size_t)h * SEQ + qbase + ml) * DK;
#pragma unroll
        for (int c = 0; c < 2; ++c)
            qf[c] = make_frag(*(const u16x8*)(qp + c * 32 + hf * 8),
                              *(const u16x8*)(qp + c * 32 + 16 + hf * 8));
    }

    float mrun[8], lrun[8];
    v8f   Oacc[4];
#pragma unroll
    for (int v = 0; v < 8; ++v) { mrun[v] = -1e30f; lrun[v] = 0.f; }
#pragma unroll
    for (int dt = 0; dt < 4; ++dt)
        Oacc[dt] = v8f{0.f, 0.f, 0.f, 0.f, 0.f, 0.f, 0.f, 0.f};

    // bias index = (k - q) + 2048, q = qbase + v + 8*hf
    const float* btab = biasTab + (size_t)h * 4096 + 2048 - qbase - 8 * hf;

    for (int jb = 0; jb < SEQ; jb += 64) {
        if (jb + 64 < SEQ) {
            __builtin_prefetch(Kb + ((size_t)h * SEQ + jb + 64 + lane) * DK, 0, 0);
            __builtin_prefetch(Vt + ((size_t)h * DK + lane * 2) * SEQ + jb + 64, 0, 0);
        }

        // ---- scores: 4 tiles of 16 kv columns, C-layout accumulators
        v8f sm[4];
#pragma unroll
        for (int t = 0; t < 4; ++t) {
            v8f s = v8f{0.f, 0.f, 0.f, 0.f, 0.f, 0.f, 0.f, 0.f};
            const u16* kp = Kb + ((size_t)h * SEQ + jb + t * 16 + ml) * DK;
#pragma unroll
            for (int c = 0; c < 2; ++c) {
                v16bf kb = make_frag(*(const u16x8*)(kp + c * 32 + hf * 8),
                                     *(const u16x8*)(kp + c * 32 + 16 + hf * 8));
                s = __builtin_amdgcn_wmma_f32_16x16x32_bf16(
                    false, qf[c], false, kb, (short)0, s, false, false);
            }
            const int kidx = jb + t * 16 + ml;
#pragma unroll
            for (int v = 0; v < 8; ++v)
                s[v] += btab[kidx - v];
            sm[t] = s;
        }

        // ---- online softmax (rows live in 16-lane groups)
        float corr[8];
#pragma unroll
        for (int v = 0; v < 8; ++v) {
            float mx = fmaxf(fmaxf(sm[0][v], sm[1][v]), fmaxf(sm[2][v], sm[3][v]));
            mx = rmax16(mx);
            float mn = fmaxf(mrun[v], mx);
            corr[v]  = __expf(mrun[v] - mn);
            mrun[v]  = mn;
        }
#pragma unroll
        for (int t = 0; t < 4; ++t)
#pragma unroll
            for (int v = 0; v < 8; ++v)
                sm[t][v] = __expf(sm[t][v] - mrun[v]);
#pragma unroll
        for (int v = 0; v < 8; ++v) {
            float rs = rsum16(sm[0][v] + sm[1][v] + sm[2][v] + sm[3][v]);
            lrun[v]  = lrun[v] * corr[v] + rs;
        }
#pragma unroll
        for (int dt = 0; dt < 4; ++dt)
#pragma unroll
            for (int v = 0; v < 8; ++v)
                Oacc[dt][v] *= corr[v];

        // ---- C-layout P -> row-major LDS (per-wave slice)
#pragma unroll
        for (int t = 0; t < 4; ++t)
#pragma unroll
            for (int v = 0; v < 8; ++v)
                Pb[wid][v + 8 * hf][t * 16 + ml] = f2bf(sm[t][v]);
        __syncthreads();

        // ---- P x V, K-chunks of 32 kv
#pragma unroll
        for (int kc = 0; kc < 2; ++kc) {
            v16bf pf = make_frag(*(const u16x8*)&Pb[wid][ml][kc * 32 + hf * 8],
                                 *(const u16x8*)&Pb[wid][ml][kc * 32 + 16 + hf * 8]);
#pragma unroll
            for (int dt = 0; dt < 4; ++dt) {
                const u16* vp = Vt + ((size_t)h * DK + dt * 16 + ml) * SEQ + jb + kc * 32;
                v16bf vf = make_frag(*(const u16x8*)(vp + hf * 8), *(const u16x8*)(vp + 16 + hf * 8));
                Oacc[dt] = __builtin_amdgcn_wmma_f32_16x16x32_bf16(
                    false, pf, false, vf, (short)0, Oacc[dt], false, false);
            }
        }
        __syncthreads();
    }

    // ---- finalize: O /= l, write bf16 to Ob[s][h*64+d]
#pragma unroll
    for (int v = 0; v < 8; ++v) {
        float inv = 1.0f / lrun[v];
        int   row = qbase + v + 8 * hf;
#pragma unroll
        for (int dt = 0; dt < 4; ++dt)
            Ob[(size_t)row * DM + h * DK + dt * 16 + ml] = f2bf(Oacc[dt][v] * inv);
    }
}

// ------------------------------------------------------------------- launcher
extern "C" void kernel_launch(void* const* d_in, const int* in_sizes, int n_in,
                              void* d_out, int out_size, void* d_ws, size_t ws_size,
                              hipStream_t stream) {
    const float* Hs = (const float*)d_in[0];
    const float* wq = (const float*)d_in[1];
    const float* wk = (const float*)d_in[2];
    const float* wv = (const float*)d_in[3];
    const float* wo = (const float*)d_in[4];
    const float* re = (const float*)d_in[5];
    float* out = (float*)d_out;

    char*        ws  = (char*)d_ws;
    const size_t MB8 = (size_t)SEQ * DM * sizeof(u16);   // 8 MiB per bf16 matrix
    u16*   Hb      = (u16*)(ws + 0 * MB8);
    u16*   Wqt     = (u16*)(ws + 1 * MB8);
    u16*   Wkt     = (u16*)(ws + 2 * MB8);
    u16*   Wvt     = (u16*)(ws + 3 * MB8);
    u16*   Wot     = (u16*)(ws + 4 * MB8);
    u16*   Qb      = (u16*)(ws + 5 * MB8);
    u16*   Kb      = (u16*)(ws + 6 * MB8);
    u16*   Vt      = (u16*)(ws + 7 * MB8);
    u16*   Ob      = (u16*)(ws + 8 * MB8);
    float* biasTab = (float*)(ws + 9 * MB8);

    cast_bf16_kernel<<<2048, 256, 0, stream>>>(Hs, Hb, SEQ * DM);

    dim3 tb(32, 8), tg(64, 64);
    tcast_kernel<<<tg, tb, 0, stream>>>(wq, Wqt);
    tcast_kernel<<<tg, tb, 0, stream>>>(wk, Wkt);
    tcast_kernel<<<tg, tb, 0, stream>>>(wv, Wvt);
    tcast_kernel<<<tg, tb, 0, stream>>>(wo, Wot);

    bias_table_kernel<<<dim3(16, NH), 256, 0, stream>>>(re, biasTab);

    dim3 gg(DM / 128, SEQ / 128);
    wmma_gemm_kernel<0><<<gg, 256, 0, stream>>>(Hb, Wqt, Qb);
    wmma_gemm_kernel<0><<<gg, 256, 0, stream>>>(Hb, Wkt, Kb);
    wmma_gemm_kernel<2><<<gg, 256, 0, stream>>>(Hb, Wvt, Vt);

    flash_attn_kernel<<<dim3(SEQ / 64, NH), 128, 0, stream>>>(Qb, Kb, Vt, biasTab, Ob);

    wmma_gemm_kernel<1><<<gg, 256, 0, stream>>>(Ob, Wot, out);
}
